// Injector_bimodal_ADA_32272384262571
// MI455X (gfx1250) — matrix operand order repaired and checked by reading, not compiled, not run on hardware
//
#include <hip/hip_runtime.h>
#include <hip/hip_bf16.h>

// ---------------------------------------------------------------------------
// Problem constants (match reference)
// ---------------------------------------------------------------------------
#define BB   2
#define LQ   4096
#define DD   768
#define NHD  6
#define NPP  4
#define HS   128
#define WS2  128
#define LF   (HS * WS2)     // 16384
#define DH   128
#define MQ   (BB * LQ)      // 8192 query rows
#define MF   (BB * LF)      // 32768 feature rows

// GEMM tiling
#define BM 64
#define BN 128
#define BK 32
#define LDA 40              // padded LDS row stride (bf16 elems) for A tile
#define LDB 40              // padded LDS row stride (bf16 elems) for B tile

typedef __attribute__((ext_vector_type(8)))  float          v8f;
typedef __attribute__((ext_vector_type(8)))  __bf16         v8bf;
typedef __attribute__((ext_vector_type(16))) __bf16         v16bf;
typedef __attribute__((ext_vector_type(4)))  unsigned short v4us;

static __device__ __forceinline__ unsigned short f2bf(float f) {
    unsigned u = __float_as_uint(f);
    unsigned r = u + 0x7FFFu + ((u >> 16) & 1u);   // round-to-nearest-even
    return (unsigned short)(r >> 16);
}
static __device__ __forceinline__ float bf2f(unsigned short u) {
    return __uint_as_float(((unsigned)u) << 16);
}

// async copy: 16 bytes global -> LDS, tracked by ASYNCcnt (CDNA5 TDM-lite path)
static __device__ __forceinline__ void async_g2l_b128(unsigned lds_off,
                                                      const void* gaddr) {
    asm volatile("global_load_async_to_lds_b128 %0, %1, off"
                 :: "v"(lds_off), "v"(gaddr) : "memory");
}
static __device__ __forceinline__ void wait_asynccnt0() {
    asm volatile("s_wait_asynccnt 0x0" ::: "memory");
}

// ---------------------------------------------------------------------------
// fp32 [K][N] -> bf16 transposed [N][K]   (weights, done once)
// ---------------------------------------------------------------------------
__global__ __launch_bounds__(256) void k_transpose_f32_bf16(
    const float* __restrict__ x, unsigned short* __restrict__ y,
    int Kdim, int Ndim) {
    int i = blockIdx.x * 256 + threadIdx.x;
    if (i < Kdim * Ndim) {
        int k = i / Ndim, n = i % Ndim;
        y[(size_t)n * Kdim + k] = f2bf(x[i]);
    }
}

// ---------------------------------------------------------------------------
// LayerNorm over last dim (cols), one row per block, bf16 output
// ---------------------------------------------------------------------------
__global__ __launch_bounds__(256) void k_layernorm_bf16(const float* __restrict__ x,
                                                        const float* __restrict__ g,
                                                        const float* __restrict__ b,
                                                        unsigned short* __restrict__ y,
                                                        int cols) {
    int row = blockIdx.x;
    const float* xr = x + (size_t)row * cols;
    int tid = threadIdx.x;
    float s = 0.f, s2 = 0.f;
    for (int c = tid; c < cols; c += 256) {
        float v = xr[c];
        s += v; s2 += v * v;
    }
    __shared__ float rs[8], rs2[8];
    for (int off = 16; off; off >>= 1) {
        s  += __shfl_down(s,  off, 32);
        s2 += __shfl_down(s2, off, 32);
    }
    int lane = tid & 31, wave = tid >> 5;
    if (lane == 0) { rs[wave] = s; rs2[wave] = s2; }
    __syncthreads();
    if (wave == 0) {
        s  = (lane < 8) ? rs[lane]  : 0.f;
        s2 = (lane < 8) ? rs2[lane] : 0.f;
        for (int off = 4; off; off >>= 1) {
            s  += __shfl_down(s,  off, 32);
            s2 += __shfl_down(s2, off, 32);
        }
        if (lane == 0) { rs[0] = s; rs2[0] = s2; }
    }
    __syncthreads();
    float mean = rs[0] / (float)cols;
    float var  = rs2[0] / (float)cols - mean * mean;
    float inv  = rsqrtf(var + 1e-6f);
    unsigned short* yr = y + (size_t)row * cols;
    for (int c = tid; c < cols; c += 256) {
        yr[c] = f2bf((xr[c] - mean) * inv * g[c] + b[c]);
    }
}

// ---------------------------------------------------------------------------
// Generic bf16 WMMA GEMM:  Y = A[M,K] @ BT[N,K]^T, templated epilogue:
//   e   = acc + bias[n] (+ addend[m,n] if ADDEND)
//   Yaux = e (if AUX)
//   out  = residual + gamma[n]*e (if RESID) else e
//   Yf32 = out (if F32), Ybf16 = bf16(out) (if B16)
// Block tile 64x128, 8 waves (4 along M x 2 along N), wave tile 16x64.
// Requires K % 32 == 0, M % 64 == 0. Double-buffered LDS staging via
// GLOBAL_LOAD_ASYNC_TO_LDS_B128 (ASYNCcnt); wave predication scalarized.
// OOB B-rows are clamped to N-1: the garbage only reaches C columns >= N,
// which the epilogue discards.
// ---------------------------------------------------------------------------
template <bool ADDEND, bool RESID, bool F32, bool B16, bool AUX>
__global__ __launch_bounds__(256) void k_gemm_bf16_wmma(
    const unsigned short* __restrict__ A,    // [M,K] bf16
    const unsigned short* __restrict__ BT,   // [N,K] bf16 (transposed weight)
    const float* __restrict__ bias,
    const float* __restrict__ addend,
    const float* __restrict__ residual,
    const float* __restrict__ gammaVec,
    float* __restrict__ Yf32,
    unsigned short* __restrict__ Ybf16,
    float* __restrict__ Yaux,
    int M, int N, int K) {

    __shared__ __align__(16) unsigned short Asm[2][BM * LDA];
    __shared__ __align__(16) unsigned short Bsm[2][BN * LDB];

    const int blockM = blockIdx.y * BM;
    const int blockN = blockIdx.x * BN;
    const int tid    = threadIdx.x;
    const int lane   = tid & 31;
    // scalarize the wave index: uniform branch + SGPR tile addressing
    const int wave   = __builtin_amdgcn_readfirstlane(tid >> 5);
    const int waveM  = (wave & 3) * 16;   // 0,16,32,48
    const int waveN  = (wave >> 2) * 64;  // 0,64
    const int half   = lane >> 4;         // lane half (0/1)
    const int l16    = lane & 15;
    const bool waveActive = (blockN + waveN) < N;

    // staging coordinates (K-contiguous for both A and BT)
    const int arow = tid >> 2;            // 0..63
    const int acol = (tid & 3) * 8;       // 0,8,16,24
    const unsigned short* aptr = A + (size_t)(blockM + arow) * K + acol;

    // B staging rows (clamped) and LDS destinations, hoisted out of the loop
    const int brow0 = tid >> 2;                 // 0..63
    const int brow1 = brow0 + 64;               // 64..127
    const int bc8   = acol;                     // 0,8,16,24
    const int gn0   = (blockN + brow0 < N) ? (blockN + brow0) : (N - 1);
    const int gn1   = (blockN + brow1 < N) ? (blockN + brow1) : (N - 1);
    const unsigned short* bptr0 = BT + (size_t)gn0 * K + bc8;
    const unsigned short* bptr1 = BT + (size_t)gn1 * K + bc8;

    v8f acc[4];
    for (int j = 0; j < 4; ++j)
        for (int i = 0; i < 8; ++i) acc[j][i] = 0.f;

    auto stage = [&](int buf, int k0) {
        async_g2l_b128((unsigned)(size_t)&Asm[buf][arow * LDA + acol], aptr + k0);
        async_g2l_b128((unsigned)(size_t)&Bsm[buf][brow0 * LDB + bc8], bptr0 + k0);
        async_g2l_b128((unsigned)(size_t)&Bsm[buf][brow1 * LDB + bc8], bptr1 + k0);
    };

    stage(0, 0);
    wait_asynccnt0();
    __syncthreads();

    int buf = 0;
    for (int k0 = 0; k0 < K; k0 += BK) {
        if (k0 + BK < K) {
            stage(buf ^ 1, k0 + BK);
            if (k0 + 2 * BK < K)
                __builtin_prefetch(aptr + k0 + 2 * BK, 0, 1);
        }
        if (waveActive) {
            const int abase = (waveM + l16) * LDA + half * 8;
            const v8bf a0 = *(const v8bf*)&Asm[buf][abase];
            const v8bf a1 = *(const v8bf*)&Asm[buf][abase + 16];
            v16bf afrag;
            for (int i = 0; i < 8; ++i) { afrag[i] = a0[i]; afrag[i + 8] = a1[i]; }
            for (int j = 0; j < 4; ++j) {
                const int nbase = (waveN + j * 16 + l16) * LDB + half * 8;
                const v8bf b0 = *(const v8bf*)&Bsm[buf][nbase];
                const v8bf b1 = *(const v8bf*)&Bsm[buf][nbase + 16];
                v16bf bfrag;
                for (int i = 0; i < 8; ++i) { bfrag[i] = b0[i]; bfrag[i + 8] = b1[i]; }
                acc[j] = __builtin_amdgcn_wmma_f32_16x16x32_bf16(
                    false, afrag, false, bfrag, (short)0, acc[j], false, false);
            }
        }
        wait_asynccnt0();          // tile k0+BK fully landed in LDS
        __syncthreads();
        buf ^= 1;
    }

    if (!waveActive) return;

    // ---- epilogue: element (m = waveM+half*8+v, n = waveN+j*16+l16) ----
    const int rowBase = blockM + waveM + half * 8;
    for (int j = 0; j < 4; ++j) {
        int col = blockN + waveN + j * 16 + l16;
        if (col >= N) continue;
        float bv = bias[col];
        float gv = 0.f;
        if constexpr (RESID) gv = gammaVec[col];
        for (int v = 0; v < 8; ++v) {
            int row = rowBase + v;
            size_t idx = (size_t)row * N + col;
            float e = acc[j][v] + bv;
            if constexpr (ADDEND) e += addend[idx];
            if constexpr (AUX)    Yaux[idx] = e;
            float o = e;
            if constexpr (RESID)  o = residual[idx] + gv * e;
            if constexpr (F32)    Yf32[idx] = o;
            if constexpr (B16)    Ybf16[idx] = f2bf(o);
        }
    }
}

// ---------------------------------------------------------------------------
// Deformable sampler: one wave per (b, q, h); lane handles 4 channels.
// value: bf16 [B][Lf][768]; off fp32 [MQ][48]; aw fp32 [MQ][24];
// refp fp32 [B][Lq][1][2]; out attn bf16 [MQ][768].
// ---------------------------------------------------------------------------
__global__ __launch_bounds__(256) void k_msdeform_sample(
    const unsigned short* __restrict__ value,
    const float* __restrict__ off,
    const float* __restrict__ aw,
    const float* __restrict__ refp,
    unsigned short* __restrict__ attn) {

    const int gw   = blockIdx.x * 8 + (threadIdx.x >> 5); // global wave id
    const int lane = threadIdx.x & 31;
    const int h = gw % NHD;
    const int r = gw / NHD;         // b*Lq + q
    const int b = r >> 12;          // Lq = 4096

    const float rx = refp[r * 2 + 0];
    const float ry = refp[r * 2 + 1];

    // softmax over the 4 points of this head
    const int abase = r * 24 + h * 4;
    float a0 = aw[abase + 0], a1 = aw[abase + 1];
    float a2 = aw[abase + 2], a3 = aw[abase + 3];
    float mx = fmaxf(fmaxf(a0, a1), fmaxf(a2, a3));
    float e0 = __expf(a0 - mx), e1 = __expf(a1 - mx);
    float e2 = __expf(a2 - mx), e3 = __expf(a3 - mx);
    float inv = 1.f / (e0 + e1 + e2 + e3);
    float wgt[4] = {e0 * inv, e1 * inv, e2 * inv, e3 * inv};

    const unsigned short* vbase =
        value + (size_t)b * LF * DD + h * DH + lane * 4;

    float acc0 = 0.f, acc1 = 0.f, acc2 = 0.f, acc3 = 0.f;
    auto corner = [&](int yy, int xx, float w) {
        if (xx >= 0 && xx < WS2 && yy >= 0 && yy < HS) {
            const v4us vv = *(const v4us*)(vbase + (size_t)(yy * WS2 + xx) * DD);
            acc0 += w * bf2f(vv[0]);
            acc1 += w * bf2f(vv[1]);
            acc2 += w * bf2f(vv[2]);
            acc3 += w * bf2f(vv[3]);
        }
    };

    const int obase = r * 48 + h * 8;
    for (int p = 0; p < NPP; ++p) {
        float lx = rx + off[obase + p * 2 + 0] * (1.f / WS2);
        float ly = ry + off[obase + p * 2 + 1] * (1.f / HS);
        float x = lx * WS2 - 0.5f;
        float y = ly * HS - 0.5f;
        float x0 = floorf(x), y0 = floorf(y);
        float wx1 = x - x0, wx0 = 1.f - wx1;
        float wy1 = y - y0, wy0 = 1.f - wy1;
        int x0i = (int)x0, y0i = (int)y0;
        float w = wgt[p];
        corner(y0i,     x0i,     wy0 * wx0 * w);
        corner(y0i,     x0i + 1, wy0 * wx1 * w);
        corner(y0i + 1, x0i,     wy1 * wx0 * w);
        corner(y0i + 1, x0i + 1, wy1 * wx1 * w);
    }

    unsigned short* op = attn + (size_t)r * DD + h * DH + lane * 4;
    v4us ov;
    ov[0] = f2bf(acc0); ov[1] = f2bf(acc1);
    ov[2] = f2bf(acc2); ov[3] = f2bf(acc3);
    *(v4us*)op = ov;
}

// ---------------------------------------------------------------------------
// Host-side launch
// ---------------------------------------------------------------------------
extern "C" void kernel_launch(void* const* d_in, const int* in_sizes, int n_in,
                              void* d_out, int out_size, void* d_ws, size_t ws_size,
                              hipStream_t stream) {
    (void)in_sizes; (void)n_in; (void)out_size; (void)ws_size;

    const float* query  = (const float*)d_in[0];
    const float* refp   = (const float*)d_in[1];
    const float* feat   = (const float*)d_in[2];
    const float* fom    = (const float*)d_in[3];
    const float* qn_g   = (const float*)d_in[6];
    const float* qn_b   = (const float*)d_in[7];
    const float* fn_g   = (const float*)d_in[8];
    const float* fn_b   = (const float*)d_in[9];
    const float* W_off  = (const float*)d_in[10];
    const float* b_off  = (const float*)d_in[11];
    const float* W_attn = (const float*)d_in[12];
    const float* b_attn = (const float*)d_in[13];
    const float* W_val  = (const float*)d_in[14];
    const float* b_val  = (const float*)d_in[15];
    const float* W_out  = (const float*)d_in[16];
    const float* b_out  = (const float*)d_in[17];
    const float* gamma  = (const float*)d_in[18];
    const float* qnm_g  = (const float*)d_in[19];
    const float* qnm_b  = (const float*)d_in[20];
    const float* fnm_g  = (const float*)d_in[21];
    const float* fnm_b  = (const float*)d_in[22];
    const float* ada_w0 = (const float*)d_in[23];
    const float* ada_b0 = (const float*)d_in[24];
    const float* ada_w1 = (const float*)d_in[25];
    const float* ada_b1 = (const float*)d_in[26];
    const float* ada_w2 = (const float*)d_in[27];
    const float* ada_b2 = (const float*)d_in[28];
    const float* g_om   = (const float*)d_in[29];

    char* ws = (char*)d_ws;
    size_t off_ = 0;
    auto carve = [&](size_t bytes) -> char* {
        off_ = (off_ + 255) & ~(size_t)255;
        char* p = ws + off_;
        off_ += bytes;
        return p;
    };

    // transposed bf16 weights: [N][K]
    unsigned short* wvalT  = (unsigned short*)carve((size_t)DD * DD * 2);
    unsigned short* woffT  = (unsigned short*)carve((size_t)48 * DD * 2);
    unsigned short* wattnT = (unsigned short*)carve((size_t)24 * DD * 2);
    unsigned short* woutT  = (unsigned short*)carve((size_t)DD * DD * 2);
    unsigned short* w0T    = (unsigned short*)carve((size_t)192 * DD * 2);
    unsigned short* w1T    = (unsigned short*)carve((size_t)192 * DD * 2);
    unsigned short* w2T    = (unsigned short*)carve((size_t)DD * 192 * 2);
    unsigned short* qn_bf    = (unsigned short*)carve((size_t)MQ * DD * 2);
    unsigned short* fn_bf    = (unsigned short*)carve((size_t)MF * DD * 2);
    unsigned short* fomn_bf  = (unsigned short*)carve((size_t)MQ * DD * 2);
    unsigned short* value_bf = (unsigned short*)carve((size_t)MF * DD * 2);
    float*          off_f32  = (float*)carve((size_t)MQ * 48 * 4);
    float*          aw_f32   = (float*)carve((size_t)MQ * 24 * 4);
    unsigned short* attn_bf  = (unsigned short*)carve((size_t)MQ * DD * 2);
    float*          q_f32    = (float*)carve((size_t)MQ * DD * 4);
    unsigned short* qn2_bf   = (unsigned short*)carve((size_t)MQ * DD * 2);
    float*          xm_f32   = (float*)carve((size_t)MQ * 192 * 4);
    unsigned short* sum_bf   = (unsigned short*)carve((size_t)MQ * 192 * 2);

    auto cvtT = [&](const float* src, unsigned short* dst, int Kd, int Nd) {
        int n = Kd * Nd;
        k_transpose_f32_bf16<<<(n + 255) / 256, 256, 0, stream>>>(src, dst, Kd, Nd);
    };
    cvtT(W_val,  wvalT,  DD, DD);
    cvtT(W_off,  woffT,  DD, 48);
    cvtT(W_attn, wattnT, DD, 24);
    cvtT(W_out,  woutT,  DD, DD);
    cvtT(ada_w0, w0T,    DD, 192);
    cvtT(ada_w1, w1T,    DD, 192);
    cvtT(ada_w2, w2T,    192, DD);

    // LayerNorms -> bf16 activations
    k_layernorm_bf16<<<MQ, 256, 0, stream>>>(query, qn_g,  qn_b,  qn_bf,   DD);
    k_layernorm_bf16<<<MF, 256, 0, stream>>>(feat,  fn_g,  fn_b,  fn_bf,   DD);
    k_layernorm_bf16<<<MQ, 256, 0, stream>>>(fom,   fnm_g, fnm_b, fomn_bf, DD);

    auto grid_of = [&](int M, int N) { return dim3((N + BN - 1) / BN, (M + BM - 1) / BM); };

    // value = LN(feat) @ W_val + b_val   (bf16 for the sampler)
    k_gemm_bf16_wmma<false, false, false, true, false>
        <<<grid_of(MF, DD), 256, 0, stream>>>(
            fn_bf, wvalT, b_val, nullptr, nullptr, nullptr,
            nullptr, value_bf, nullptr, MF, DD, DD);
    // sampling offsets & attention logits (fp32)
    k_gemm_bf16_wmma<false, false, true, false, false>
        <<<grid_of(MQ, 48), 256, 0, stream>>>(
            qn_bf, woffT, b_off, nullptr, nullptr, nullptr,
            off_f32, nullptr, nullptr, MQ, 48, DD);
    k_gemm_bf16_wmma<false, false, true, false, false>
        <<<grid_of(MQ, 24), 256, 0, stream>>>(
            qn_bf, wattnT, b_attn, nullptr, nullptr, nullptr,
            aw_f32, nullptr, nullptr, MQ, 24, DD);

    // deformable bilinear sampling -> attn (bf16)
    k_msdeform_sample<<<(MQ * NHD) / 8, 256, 0, stream>>>(
        value_bf, off_f32, aw_f32, refp, attn_bf);

    // q = query + gamma * (attn @ W_out + b_out)
    k_gemm_bf16_wmma<false, true, true, false, false>
        <<<grid_of(MQ, DD), 256, 0, stream>>>(
            attn_bf, woutT, b_out, nullptr, query, gamma,
            q_f32, nullptr, nullptr, MQ, DD, DD);

    // qn2 = LN(q)
    k_layernorm_bf16<<<MQ, 256, 0, stream>>>(q_f32, qnm_g, qnm_b, qn2_bf, DD);

    // xm = LN(fom) @ ada_w1 + ada_b1  (fp32)
    k_gemm_bf16_wmma<false, false, true, false, false>
        <<<grid_of(MQ, 192), 256, 0, stream>>>(
            fomn_bf, w1T, ada_b1, nullptr, nullptr, nullptr,
            xm_f32, nullptr, nullptr, MQ, 192, DD);
    // sum = bf16( qn2 @ ada_w0 + ada_b0 + xm )
    k_gemm_bf16_wmma<true, false, false, true, false>
        <<<grid_of(MQ, 192), 256, 0, stream>>>(
            qn2_bf, w0T, ada_b0, xm_f32, nullptr, nullptr,
            nullptr, sum_bf, nullptr, MQ, 192, DD);

    // prompted = sum @ ada_w2 + ada_b2 ;  q_out = q + gamma_om * prompted
    float* out_q        = (float*)d_out;
    float* out_prompted = (float*)d_out + (size_t)MQ * DD;
    k_gemm_bf16_wmma<false, true, true, false, true>
        <<<grid_of(MQ, DD), 256, 0, stream>>>(
            sum_bf, w2T, ada_b2, nullptr, q_f32, g_om,
            out_q, nullptr, out_prompted, MQ, DD, 192);
}
